// TemporalCausalModel_32392643346797
// MI455X (gfx1250) — matrix operand order repaired and checked by reading, not compiled, not run on hardware
//
#include <hip/hip_runtime.h>
#include <stdint.h>

// ---------------- constants (match reference) ----------------
enum {
  NNODES = 100000, NINC = 400000, NEDGES = 20000, NGRAPHS = 512,
  NSEQ = 16, SEQLEN = 32, FIN = 128, HID = 256, HEADS = 4, ODIM = 64,
  RNNH = 256, GATES = 4 * RNNH, OUTDIM = 128
};

typedef __attribute__((ext_vector_type(8)))  __bf16 v8bf;
typedef __attribute__((ext_vector_type(16))) __bf16 v16bf;
typedef __attribute__((ext_vector_type(8)))  float  v8f;

// f32 -> bf16 round-to-nearest-even
__device__ __forceinline__ unsigned short f2bf(float f) {
  unsigned int u = __float_as_uint(f);
  unsigned int r = u + 0x7FFFu + ((u >> 16) & 1u);
  return (unsigned short)(r >> 16);
}

__device__ __forceinline__ void atomicMaxF(float* addr, float val) {
  int* ai = (int*)addr;
  int old = __float_as_int(*addr);
  while (__int_as_float(old) < val) {
    int assumed = old;
    old = atomicCAS(ai, assumed, __float_as_int(val));
    if (old == assumed) break;
  }
}

// ---------------- generic small kernels ----------------
__global__ void k_fill_f32(float* p, float v, long long n) {
  long long i = (long long)blockIdx.x * blockDim.x + threadIdx.x;
  if (i < n) p[i] = v;
}

__global__ void k_f32_to_bf16(const float* in, unsigned short* out, long long n) {
  long long i = (long long)blockIdx.x * blockDim.x + threadIdx.x;
  if (i < n) out[i] = f2bf(in[i]);
}

// in: [K][N] row-major f32  ->  out: [N][K] row-major bf16
__global__ void k_transpose_cvt(const float* in, unsigned short* out, int K, int N) {
  long long i = (long long)blockIdx.x * blockDim.x + threadIdx.x;
  if (i >= (long long)K * N) return;
  int k = (int)(i / N), n = (int)(i - (long long)k * N);
  out[(long long)n * K + k] = f2bf(in[i]);
}

__global__ void k_vec_add(const float* a, const float* b, float* c, int n) {
  int i = blockIdx.x * blockDim.x + threadIdx.x;
  if (i < n) c[i] = a[i] + b[i];
}

// ---------------- WMMA bf16 GEMM: C[M,N] = A[M,K] * B^T (B stored [N,K]) ----
// Register-blocked: each wave computes 1 M-tile x NT N-tiles; the A fragment is
// loaded once per k-chunk and reused across NT WMMAs. K is a template constant
// so every load in the (fully unrolled) K loop uses an immediate offset from a
// single base pointer: no address VALU inside the loop, no WMMA->VALU hazard
// nops, and the scheduler can clause-batch loads against WMMAs.
// EPI: 0 = none, 1 = +bias, 2 = +bias,relu, 3 = +bias,sigmoid
#define GEMM_NT 8

template <int EPI, int K>
__global__ void k_gemm_wmma(const unsigned short* A_, const unsigned short* B_,
                            const float* bias, float* C,
                            int M, int N, int total_waves) {
  const __bf16* A = (const __bf16*)A_;
  const __bf16* B = (const __bf16*)B_;
  int lane = threadIdx.x & 31;
  int wid = blockIdx.x * (blockDim.x >> 5) + (threadIdx.x >> 5);
  if (wid >= total_waves) return;               // wave-uniform exit
  int ngrp = (N >> 4) / GEMM_NT;                // N-tile groups per M row
  int tm = wid / ngrp;
  int tg = wid - tm * ngrp;
  // A operand (16x32 bf16): lane l -> row m=l%16, chunks k0=(l/16)*8 and k0+16
  const __bf16* a_row = A + (long long)(tm * 16 + (lane & 15)) * K + ((lane >> 4) * 8);
  // B operand (32x16 bf16): lane l -> col n=l%16, chunks k0=(l/16)*8 and k0+16
  const __bf16* b_base = B + (long long)(tg * GEMM_NT * 16 + (lane & 15)) * K + ((lane >> 4) * 8);

  v8f acc[GEMM_NT];
#pragma unroll
  for (int j = 0; j < GEMM_NT; ++j)
    acc[j] = (v8f){0.f, 0.f, 0.f, 0.f, 0.f, 0.f, 0.f, 0.f};

#pragma unroll
  for (int k = 0; k < K; k += 32) {
    v8bf alo = *(const v8bf*)(a_row + k);
    v8bf ahi = *(const v8bf*)(a_row + k + 16);
    v16bf av = __builtin_shufflevector(alo, ahi, 0, 1, 2, 3, 4, 5, 6, 7,
                                       8, 9, 10, 11, 12, 13, 14, 15);
    v16bf bv[GEMM_NT];
#pragma unroll
    for (int j = 0; j < GEMM_NT; ++j)
      bv[j] = __builtin_shufflevector(*(const v8bf*)(b_base + j * 16 * K + k), *(const v8bf*)(b_base + j * 16 * K + k + 16), 0, 1, 2, 3, 4, 5, 6, 7, 8, 9, 10, 11, 12, 13, 14, 15);   // imm offsets
#pragma unroll
    for (int j = 0; j < GEMM_NT; ++j)
      acc[j] = __builtin_amdgcn_wmma_f32_16x16x32_bf16(false, av, false, bv[j],
                                                       (short)0, acc[j], false, false);
  }
  // C layout: VGPR r, lane l -> M = r + 8*(l/16), N = l%16
  int mb = tm * 16 + ((lane >> 4) << 3);
#pragma unroll
  for (int j = 0; j < GEMM_NT; ++j) {
    int n0 = (tg * GEMM_NT + j) * 16 + (lane & 15);
    float bval = (EPI > 0) ? bias[n0] : 0.f;
#pragma unroll
    for (int r = 0; r < 8; ++r) {
      float v = acc[j][r] + bval;
      if (EPI == 2) v = fmaxf(v, 0.f);
      if (EPI == 3) v = 1.f / (1.f + __expf(-v));
      C[(long long)(mb + r) * N + n0] = v;
    }
  }
}

// ---------------- hypergraph kernels ----------------
__global__ void k_counts(const int* he_node, const int* he_edge,
                         float* cnt_e, float* deg, int n) {
  int i = blockIdx.x * blockDim.x + threadIdx.x;
  if (i >= n) return;
  atomicAdd(&cnt_e[he_edge[i]], 1.f);
  atomicAdd(&deg[he_node[i]], 1.f);
}

__global__ void k_eattr_acc(const float* x, const int* he_node, const int* he_edge,
                            float* eattr, long long n /* NINC*FIN */) {
  long long i = (long long)blockIdx.x * blockDim.x + threadIdx.x;
  if (i >= n) return;
  long long inc = i >> 7;           // /128
  int f = (int)(i & 127);
  atomicAdd(&eattr[(long long)he_edge[inc] * FIN + f],
            x[(long long)he_node[inc] * FIN + f]);
}

__global__ void k_eattr_fin(float* eattr, const float* cnt_e,
                            unsigned short* ebf, long long n) {
  long long i = (long long)blockIdx.x * blockDim.x + threadIdx.x;
  if (i >= n) return;
  int e = (int)(i >> 7);
  float v = eattr[i] / fmaxf(cnt_e[e], 1.f);
  eattr[i] = v;
  ebf[i] = f2bf(v);
}

// s[row][h] = dot(feat[row][h*64 : h*64+64], att[h][att_off : att_off+64])
__global__ void k_scores(const float* feat, const float* att, int att_off,
                         float* s, int rows) {
  long long i = (long long)blockIdx.x * blockDim.x + threadIdx.x;
  if (i >= (long long)rows * HEADS) return;
  int row = (int)(i >> 2), h = (int)(i & 3);
  const float* f = feat + (long long)row * HID + h * ODIM;
  const float* a = att + h * 2 * ODIM + att_off;
  float acc = 0.f;
#pragma unroll 8
  for (int k = 0; k < ODIM; ++k) acc += f[k] * a[k];
  s[i] = acc;
}

__global__ void k_logits_max(const float* sn, const float* se,
                             const int* he_node, const int* he_edge,
                             float* ebuf, float* medge, long long n /* NINC*4 */) {
  long long i = (long long)blockIdx.x * blockDim.x + threadIdx.x;
  if (i >= n) return;
  long long inc = i >> 2;
  int h = (int)(i & 3);
  float l = sn[(long long)he_node[inc] * HEADS + h] +
            se[(long long)he_edge[inc] * HEADS + h];
  l = (l > 0.f) ? l : 0.2f * l;  // leaky relu
  ebuf[i] = l;
  atomicMaxF(&medge[(long long)he_edge[inc] * HEADS + h], l);
}

__global__ void k_expsum(const int* he_edge, float* ebuf, const float* medge,
                         float* sume, long long n) {
  long long i = (long long)blockIdx.x * blockDim.x + threadIdx.x;
  if (i >= n) return;
  long long inc = i >> 2;
  int h = (int)(i & 3);
  long long ei = (long long)he_edge[inc] * HEADS + h;
  float v = __expf(ebuf[i] - medge[ei]);
  ebuf[i] = v;
  atomicAdd(&sume[ei], v);
}

__global__ void k_oute_acc(const float* xh, const int* he_node, const int* he_edge,
                           const float* ebuf, const float* sume, float* oute,
                           long long n /* NINC*HID */) {
  long long i = (long long)blockIdx.x * blockDim.x + threadIdx.x;
  if (i >= n) return;
  long long inc = i >> 8;
  int c = (int)(i & 255), h = c >> 6;
  int e = he_edge[inc];
  float a = ebuf[inc * 4 + h] / fmaxf(sume[(long long)e * 4 + h], 1e-16f);
  atomicAdd(&oute[(long long)e * HID + c],
            a * xh[(long long)he_node[inc] * HID + c]);
}

__global__ void k_oute_scale(float* oute, const float* cnt_e, long long n) {
  long long i = (long long)blockIdx.x * blockDim.x + threadIdx.x;
  if (i >= n) return;
  float c = cnt_e[i >> 8];
  oute[i] *= (c > 0.f) ? 1.f / c : 0.f;
}

__global__ void k_outn_acc(const float* oute, const int* he_node, const int* he_edge,
                           const float* ebuf, const float* sume, float* hout,
                           long long n /* NINC*HID */) {
  long long i = (long long)blockIdx.x * blockDim.x + threadIdx.x;
  if (i >= n) return;
  long long inc = i >> 8;
  int c = (int)(i & 255), h = c >> 6;
  int e = he_edge[inc];
  float a = ebuf[inc * 4 + h] / fmaxf(sume[(long long)e * 4 + h], 1e-16f);
  atomicAdd(&hout[(long long)he_node[inc] * HID + c],
            a * oute[(long long)e * HID + c]);
}

// h = relu(Dinv*acc + bias); pooled[batch] += h
__global__ void k_node_fin(float* hout, const float* deg, const float* bias,
                           const int* batch, float* pooled, long long n /* NNODES*HID */) {
  long long i = (long long)blockIdx.x * blockDim.x + threadIdx.x;
  if (i >= n) return;
  int node = (int)(i >> 8), c = (int)(i & 255);
  float d = deg[node];
  float v = fmaxf((d > 0.f ? 1.f / d : 0.f) * hout[i] + bias[c], 0.f);
  hout[i] = v;
  atomicAdd(&pooled[(long long)batch[node] * HID + c], v);
}

__global__ void k_count_graph(const int* batch, float* cntg, int n) {
  int i = blockIdx.x * blockDim.x + threadIdx.x;
  if (i < n) atomicAdd(&cntg[batch[i]], 1.f);
}

__global__ void k_pool_fin(float* pooled, const float* cntg, unsigned short* pbf) {
  int i = blockIdx.x * blockDim.x + threadIdx.x;
  if (i >= NGRAPHS * HID) return;
  float v = pooled[i] / fmaxf(cntg[i >> 8], 1.f);
  pooled[i] = v;
  pbf[i] = f2bf(v);
}

// ---------------- LSTM recurrence (one block per direction) ----------------
// Gin: [2][512][1024] input projections (bias already folded), w_hh: [2][1024][256]
// Y:   [512][512], column offset dir*256. B(=NSEQ)=16 sequences, T=32 steps.
__global__ void __launch_bounds__(1024)
k_lstm_rec(const float* Gin, const float* w_hh, float* Y) {
  int dir = blockIdx.x;
  const float* gin = Gin + (long long)dir * NGRAPHS * GATES;
  const float* whh = w_hh + (long long)dir * GATES * RNNH;
  __shared__ float h_sh[NSEQ * RNNH];     // 16 KB
  __shared__ float g_sh[NSEQ * GATES];    // 64 KB
  int tid = threadIdx.x;                  // 1024 threads
  for (int i = tid; i < NSEQ * RNNH; i += 1024) h_sh[i] = 0.f;
  float c_reg[4] = {0.f, 0.f, 0.f, 0.f};
  __syncthreads();
  for (int step = 0; step < SEQLEN; ++step) {
    int t = dir ? (SEQLEN - 1 - step) : step;
    // phase 1: gate row r = tid, dot with h for all 16 sequences
    {
      const float* wrow = whh + (long long)tid * RNNH;
      float acc[NSEQ];
#pragma unroll
      for (int b = 0; b < NSEQ; ++b) acc[b] = 0.f;
      for (int k = 0; k < RNNH; k += 4) {
        float w0 = wrow[k], w1 = wrow[k + 1], w2 = wrow[k + 2], w3 = wrow[k + 3];
#pragma unroll
        for (int b = 0; b < NSEQ; ++b) {
          const float* hb = h_sh + b * RNNH + k;
          acc[b] += hb[0] * w0 + hb[1] * w1 + hb[2] * w2 + hb[3] * w3;
        }
      }
#pragma unroll
      for (int b = 0; b < NSEQ; ++b)
        g_sh[b * GATES + tid] = acc[b] + gin[(long long)(b * SEQLEN + t) * GATES + tid];
    }
    __syncthreads();
    // phase 2: state update, 4 cells per thread (torch gate order i,f,g,o)
#pragma unroll
    for (int q = 0; q < 4; ++q) {
      int cell = tid * 4 + q;            // 0..4095
      int b = cell >> 8, j = cell & 255;
      float gi = g_sh[b * GATES + j];
      float gf = g_sh[b * GATES + RNNH + j];
      float gg = g_sh[b * GATES + 2 * RNNH + j];
      float go = g_sh[b * GATES + 3 * RNNH + j];
      float is = 1.f / (1.f + __expf(-gi));
      float fs = 1.f / (1.f + __expf(-gf));
      float os = 1.f / (1.f + __expf(-go));
      float c = fs * c_reg[q] + is * tanhf(gg);
      c_reg[q] = c;
      float h = os * tanhf(c);
      h_sh[b * RNNH + j] = h;
      Y[(long long)(b * SEQLEN + t) * (2 * RNNH) + dir * RNNH + j] = h;
    }
    __syncthreads();
  }
}

// ---------------- host side ----------------
static inline dim3 flat_grid(long long n, int bs) {
  return dim3((unsigned)((n + bs - 1) / bs));
}

template <int EPI, int K>
static void launch_gemm(const unsigned short* A, const unsigned short* B,
                        const float* bias, float* C, int M, int N,
                        hipStream_t s) {
  int total_waves = (M / 16) * ((N / 16) / GEMM_NT);
  k_gemm_wmma<EPI, K><<<dim3((total_waves + 7) / 8), dim3(256), 0, s>>>(A, B, bias, C, M, N, total_waves);
}

extern "C" void kernel_launch(void* const* d_in, const int* in_sizes, int n_in,
                              void* d_out, int out_size, void* d_ws, size_t ws_size,
                              hipStream_t stream) {
  (void)in_sizes; (void)n_in; (void)out_size; (void)ws_size;
  const float* x        = (const float*)d_in[0];
  const int*   he_node  = (const int*)d_in[1];
  const int*   he_edge  = (const int*)d_in[2];
  const int*   batch    = (const int*)d_in[3];
  const float* w_lin    = (const float*)d_in[4];
  const float* att      = (const float*)d_in[5];
  const float* bias     = (const float*)d_in[6];
  const float* w_ih0    = (const float*)d_in[7];
  const float* w_hh0    = (const float*)d_in[8];
  const float* b_ih0    = (const float*)d_in[9];
  const float* b_hh0    = (const float*)d_in[10];
  const float* w_ih1    = (const float*)d_in[11];
  const float* w_hh1    = (const float*)d_in[12];
  const float* b_ih1    = (const float*)d_in[13];
  const float* b_hh1    = (const float*)d_in[14];
  const float* w_causal = (const float*)d_in[15];
  const float* b_causal = (const float*)d_in[16];
  const float* w_link   = (const float*)d_in[17];
  const float* b_link   = (const float*)d_in[18];
  float* out = (float*)d_out;

  // -------- workspace carve (256B aligned chunks) --------
  char* p = (char*)d_ws;
  auto take = [&](size_t bytes) -> void* {
    void* r = (void*)p;
    p += (bytes + 255) & ~(size_t)255;
    return r;
  };
  unsigned short* xbf     = (unsigned short*)take((size_t)NNODES * FIN * 2);
  unsigned short* wlinT   = (unsigned short*)take((size_t)HID * FIN * 2);
  float* xh               = (float*)take((size_t)NNODES * HID * 4);   // aliased as hout later
  float* cnt_e            = (float*)take((size_t)NEDGES * 4);
  float* deg              = (float*)take((size_t)NNODES * 4);
  float* eattr            = (float*)take((size_t)NEDGES * FIN * 4);
  unsigned short* eattrbf = (unsigned short*)take((size_t)NEDGES * FIN * 2);
  float* eh               = (float*)take((size_t)NEDGES * HID * 4);
  float* s_node           = (float*)take((size_t)NNODES * HEADS * 4);
  float* s_edge           = (float*)take((size_t)NEDGES * HEADS * 4);
  float* medge            = (float*)take((size_t)NEDGES * HEADS * 4);
  float* sume             = (float*)take((size_t)NEDGES * HEADS * 4);
  float* ebuf             = (float*)take((size_t)NINC * HEADS * 4);
  float* oute             = (float*)take((size_t)NEDGES * HID * 4);
  float* pooled           = (float*)take((size_t)NGRAPHS * HID * 4);
  float* cntg             = (float*)take((size_t)NGRAPHS * 4);
  unsigned short* poolbf  = (unsigned short*)take((size_t)NGRAPHS * HID * 2);
  unsigned short* wih0bf  = (unsigned short*)take((size_t)2 * GATES * HID * 2);
  unsigned short* wih1bf  = (unsigned short*)take((size_t)2 * GATES * 2 * RNNH * 2);
  float* bsum0            = (float*)take((size_t)2 * GATES * 4);
  float* bsum1            = (float*)take((size_t)2 * GATES * 4);
  float* gin0             = (float*)take((size_t)2 * NGRAPHS * GATES * 4);
  float* y1               = (float*)take((size_t)NGRAPHS * 2 * RNNH * 4);
  unsigned short* y1bf    = (unsigned short*)take((size_t)NGRAPHS * 2 * RNNH * 2);
  float* gin1             = (float*)take((size_t)2 * NGRAPHS * GATES * 4);
  float* y2               = (float*)take((size_t)NGRAPHS * 2 * RNNH * 4);
  unsigned short* y2bf    = (unsigned short*)take((size_t)NGRAPHS * 2 * RNNH * 2);
  unsigned short* wcausT  = (unsigned short*)take((size_t)HID * 2 * RNNH * 2);
  float* ccaus            = (float*)take((size_t)NGRAPHS * HID * 4);
  unsigned short* ccausbf = (unsigned short*)take((size_t)NGRAPHS * HID * 2);
  unsigned short* wlinkT  = (unsigned short*)take((size_t)OUTDIM * HID * 2);
  float* hout = xh;  // reuse of the 102 MB buffer once xh is dead

  const int BS = 256;

  // -------- init accumulators --------
  hipMemsetAsync(cnt_e, 0, (size_t)NEDGES * 4, stream);
  hipMemsetAsync(deg, 0, (size_t)NNODES * 4, stream);
  hipMemsetAsync(eattr, 0, (size_t)NEDGES * FIN * 4, stream);
  hipMemsetAsync(sume, 0, (size_t)NEDGES * HEADS * 4, stream);
  hipMemsetAsync(oute, 0, (size_t)NEDGES * HID * 4, stream);
  hipMemsetAsync(pooled, 0, (size_t)NGRAPHS * HID * 4, stream);
  hipMemsetAsync(cntg, 0, (size_t)NGRAPHS * 4, stream);
  k_fill_f32<<<flat_grid(NEDGES * HEADS, BS), BS, 0, stream>>>(medge, -3e38f, NEDGES * HEADS);

  // -------- weight / input preparation (bf16) --------
  k_f32_to_bf16<<<flat_grid((long long)NNODES * FIN, BS), BS, 0, stream>>>(x, xbf, (long long)NNODES * FIN);
  k_transpose_cvt<<<flat_grid((long long)FIN * HID, BS), BS, 0, stream>>>(w_lin, wlinT, FIN, HID);
  k_f32_to_bf16<<<flat_grid((long long)2 * GATES * HID, BS), BS, 0, stream>>>(w_ih0, wih0bf, (long long)2 * GATES * HID);
  k_f32_to_bf16<<<flat_grid((long long)2 * GATES * 2 * RNNH, BS), BS, 0, stream>>>(w_ih1, wih1bf, (long long)2 * GATES * 2 * RNNH);
  k_transpose_cvt<<<flat_grid((long long)2 * RNNH * HID, BS), BS, 0, stream>>>(w_causal, wcausT, 2 * RNNH, HID);
  k_transpose_cvt<<<flat_grid((long long)HID * OUTDIM, BS), BS, 0, stream>>>(w_link, wlinkT, HID, OUTDIM);
  k_vec_add<<<flat_grid(2 * GATES, BS), BS, 0, stream>>>(b_ih0, b_hh0, bsum0, 2 * GATES);
  k_vec_add<<<flat_grid(2 * GATES, BS), BS, 0, stream>>>(b_ih1, b_hh1, bsum1, 2 * GATES);

  // -------- hypergraph conv --------
  // xh = x @ w_lin   (100000 x 128 x 256, WMMA bf16)
  launch_gemm<0, FIN>(xbf, wlinT, nullptr, xh, NNODES, HID, stream);
  k_counts<<<flat_grid(NINC, BS), BS, 0, stream>>>(he_node, he_edge, cnt_e, deg, NINC);
  k_eattr_acc<<<flat_grid((long long)NINC * FIN, BS), BS, 0, stream>>>(x, he_node, he_edge, eattr, (long long)NINC * FIN);
  k_eattr_fin<<<flat_grid((long long)NEDGES * FIN, BS), BS, 0, stream>>>(eattr, cnt_e, eattrbf, (long long)NEDGES * FIN);
  // eh = eattr @ w_lin  (20000 x 128 x 256)
  launch_gemm<0, FIN>(eattrbf, wlinT, nullptr, eh, NEDGES, HID, stream);
  // per-node / per-edge attention scores (logit decomposition)
  k_scores<<<flat_grid((long long)NNODES * HEADS, BS), BS, 0, stream>>>(xh, att, 0, s_node, NNODES);
  k_scores<<<flat_grid((long long)NEDGES * HEADS, BS), BS, 0, stream>>>(eh, att, ODIM, s_edge, NEDGES);
  // segment softmax over hyperedges
  k_logits_max<<<flat_grid((long long)NINC * HEADS, BS), BS, 0, stream>>>(s_node, s_edge, he_node, he_edge, ebuf, medge, (long long)NINC * HEADS);
  k_expsum<<<flat_grid((long long)NINC * HEADS, BS), BS, 0, stream>>>(he_edge, ebuf, medge, sume, (long long)NINC * HEADS);
  // node -> hyperedge
  k_oute_acc<<<flat_grid((long long)NINC * HID, BS), BS, 0, stream>>>(xh, he_node, he_edge, ebuf, sume, oute, (long long)NINC * HID);
  k_oute_scale<<<flat_grid((long long)NEDGES * HID, BS), BS, 0, stream>>>(oute, cnt_e, (long long)NEDGES * HID);
  // hyperedge -> node (xh is dead now; reuse its buffer as hout)
  hipMemsetAsync(hout, 0, (size_t)NNODES * HID * 4, stream);
  k_outn_acc<<<flat_grid((long long)NINC * HID, BS), BS, 0, stream>>>(oute, he_node, he_edge, ebuf, sume, hout, (long long)NINC * HID);
  // finalize + fused mean-pool accumulation
  k_node_fin<<<flat_grid((long long)NNODES * HID, BS), BS, 0, stream>>>(hout, deg, bias, batch, pooled, (long long)NNODES * HID);
  k_count_graph<<<flat_grid(NNODES, BS), BS, 0, stream>>>(batch, cntg, NNODES);
  k_pool_fin<<<flat_grid(NGRAPHS * HID, BS), BS, 0, stream>>>(pooled, cntg, poolbf);

  // -------- BiLSTM layer 0 --------
  for (int d = 0; d < 2; ++d)
    launch_gemm<1, HID>(poolbf, wih0bf + (size_t)d * GATES * HID, bsum0 + d * GATES,
                        gin0 + (size_t)d * NGRAPHS * GATES, NGRAPHS, GATES, stream);
  k_lstm_rec<<<dim3(2), dim3(1024), 0, stream>>>(gin0, w_hh0, y1);
  k_f32_to_bf16<<<flat_grid((long long)NGRAPHS * 2 * RNNH, BS), BS, 0, stream>>>(y1, y1bf, (long long)NGRAPHS * 2 * RNNH);

  // -------- BiLSTM layer 1 --------
  for (int d = 0; d < 2; ++d)
    launch_gemm<1, 2 * RNNH>(y1bf, wih1bf + (size_t)d * GATES * 2 * RNNH, bsum1 + d * GATES,
                             gin1 + (size_t)d * NGRAPHS * GATES, NGRAPHS, GATES, stream);
  k_lstm_rec<<<dim3(2), dim3(1024), 0, stream>>>(gin1, w_hh1, y2);
  k_f32_to_bf16<<<flat_grid((long long)NGRAPHS * 2 * RNNH, BS), BS, 0, stream>>>(y2, y2bf, (long long)NGRAPHS * 2 * RNNH);

  // -------- head: causal = relu(l2 @ w_causal + b), out = sigmoid(causal @ w_link + b) --------
  launch_gemm<2, 2 * RNNH>(y2bf, wcausT, b_causal, ccaus, NGRAPHS, HID, stream);
  k_f32_to_bf16<<<flat_grid((long long)NGRAPHS * HID, BS), BS, 0, stream>>>(ccaus, ccausbf, (long long)NGRAPHS * HID);
  launch_gemm<3, HID>(ccausbf, wlinkT, b_link, out, NGRAPHS, OUTDIM, stream);
}
